// EdgeTransformerLayer_23991687315684
// MI455X (gfx1250) — compile-verified, matrix operand order checked by reading, not run on hardware
//
#include <hip/hip_runtime.h>

// ---------------------------------------------------------------------------
// EdgeTransformerLayer on gfx1250 (MI455X).
// bf16 WMMA for all GEMMs; TDM (tensor_load_to_lds) + double-buffered LDS
// staging for the big GEMMs; TENSORcnt/barrier sync per CDNA5 ISA.
// B=2, N=64, D=256, H=8, DK=32, DFF=1024, M = B*N*N = 8192.
// ---------------------------------------------------------------------------

typedef __attribute__((ext_vector_type(16))) __bf16        v16bf;
typedef __attribute__((ext_vector_type(8)))  float         v8f;
typedef __attribute__((ext_vector_type(4)))  unsigned int  u32x4;
typedef __attribute__((ext_vector_type(8)))  int           i32x8;
typedef __attribute__((ext_vector_type(4)))  int           i32x4;

constexpr int kB   = 2;
constexpr int kN   = 64;
constexpr int kD   = 256;
constexpr int kH   = 8;
constexpr int kDK  = 32;
constexpr int kDFF = 1024;
constexpr int kM   = kB * kN * kN;          // 8192 rows
constexpr float kEPS = 1e-5f;
constexpr float kNEG = -1e9f;
constexpr float kSCALE = 0.17677669529663687f; // 1/sqrt(32)

// LDS tile geometry: rows of 64 bf16 (128B) padded by TDM with 4 dwords to a
// 144B pitch (=36 banks) so 16-lane ds_load_b128 fragment reads spread banks.
constexpr int kRowPitch   = 72;                    // elements (144 bytes)
constexpr int kAtileElems = 128 * kRowPitch;       // 18432 B
constexpr int kBtileElems = 64  * kRowPitch;       //  9216 B
constexpr int kBufElems   = kAtileElems + kBtileElems; // 27648 B per buffer

// ---- fp32 <-> bf16 helpers (round-to-nearest-even) ------------------------
__device__ __forceinline__ unsigned short f2bf(float f) {
  unsigned int u = __float_as_uint(f);
  u += 0x7FFFu + ((u >> 16) & 1u);
  return (unsigned short)(u >> 16);
}
__device__ __forceinline__ float bf2f(unsigned short s) {
  return __uint_as_float(((unsigned int)s) << 16);
}

union FragBF { v16bf v; unsigned int u[8]; };
union AccF   { v8f   v; float f[8]; };

// 16x32 bf16 WMMA A/B-fragment from a row-major matrix (ISA §7.12.2 layout):
//  lanes 0-15 : row = lane,    K {0..7}(v0-3) {16..23}(v4-7)
//  lanes 16-31: row = lane-16, K {8..15}      {24..31}
// -> per lane two contiguous 16B chunks at +0 and +32 bytes.
__device__ __forceinline__ v16bf load_frag(const unsigned short* base,
                                           int row_stride, int r0, int k0) {
  int lane = threadIdx.x & 31;
  const unsigned short* p =
      base + (size_t)(r0 + (lane & 15)) * row_stride + k0 + ((lane & 16) ? 8 : 0);
  FragBF fr;
#pragma unroll
  for (int e2 = 0; e2 < 8; ++e2) {
    int k = ((e2 & 3) * 2) + ((e2 >> 2) << 4);
    fr.u[e2] = *(const unsigned int*)(p + k);
  }
  return fr.v;
}

// Same fragment gather but from the padded LDS tile (pitch 72 elements).
__device__ __forceinline__ v16bf lds_frag(const unsigned short* sb, int r0, int k0) {
  int lane = threadIdx.x & 31;
  const unsigned short* p =
      sb + (r0 + (lane & 15)) * kRowPitch + k0 + ((lane & 16) ? 8 : 0);
  FragBF fr;
#pragma unroll
  for (int e2 = 0; e2 < 8; ++e2) {
    int k = ((e2 & 3) * 2) + ((e2 >> 2) << 4);
    fr.u[e2] = *(const unsigned int*)(p + k);
  }
  return fr.v;
}

// ---- TDM: 2-D tile DMA global -> LDS (ISA 08_async_tensor §8) -------------
// data_size=2B; tile_dim0 = 64 elems/row; LDS padding 4 dwords per 32 dwords.
__device__ __forceinline__ void tdm_load_2d(const unsigned short* gaddr,
                                            unsigned lds_addr, int tile_rows,
                                            unsigned tensor_k, unsigned tensor_rows,
                                            unsigned row_stride_elems) {
  unsigned long long ga = (unsigned long long)(size_t)gaddr;
  u32x4 g0;
  g0[0] = 1u;                                   // count=1 (valid), user mode
  g0[1] = lds_addr;                             // LDS byte address
  g0[2] = (unsigned)(ga & 0xFFFFFFFFu);         // global_addr[31:0]
  g0[3] = (unsigned)((ga >> 32) & 0x01FFFFFFu)  // global_addr[56:32]
          | (2u << 30);                         // type=2 ("image")
  i32x8 g1;
  unsigned w0 = (1u << 16)                      // data_size=1 -> 2 bytes
              | (1u << 20)                      // pad_enable
              | (4u << 22)                      // pad_interval: 32 dwords
              | (3u << 25);                     // pad_amount: 4 dwords
  g1[0] = (int)w0;
  g1[1] = (int)((tensor_k & 0xFFFFu) << 16);    // tensor_dim0[15:0]
  g1[2] = (int)(((tensor_k >> 16) & 0xFFFFu) | ((tensor_rows & 0xFFFFu) << 16));
  g1[3] = (int)(((tensor_rows >> 16) & 0xFFFFu) | (64u << 16)); // tile_dim0=64
  g1[4] = (int)(tile_rows & 0xFFFF);            // tile_dim1 (tile_dim2=0)
  g1[5] = (int)row_stride_elems;                // tensor_dim0_stride[31:0]
  g1[6] = 0;
  g1[7] = 0;
  i32x4 z4 = {0, 0, 0, 0};
#if defined(__clang_major__) && __clang_major__ >= 23
  i32x8 z8 = {0, 0, 0, 0, 0, 0, 0, 0};
  __builtin_amdgcn_tensor_load_to_lds(g0, g1, z4, z4, z8, 0);
#else
  __builtin_amdgcn_tensor_load_to_lds(g0, g1, z4, z4, 0);
#endif
}

__device__ __forceinline__ void wait_tensorcnt0() {
#if __has_builtin(__builtin_amdgcn_s_wait_tensorcnt)
  __builtin_amdgcn_s_wait_tensorcnt(0);
#else
  asm volatile("s_wait_tensorcnt 0x0" ::: "memory");
#endif
}

// ---- weight convert + transpose: fp32 [K,N] -> bf16 [N,K] -----------------
__global__ void wconv_t(const float* __restrict__ W, unsigned short* __restrict__ Wt,
                        int K, int Ncol) {
  int idx = blockIdx.x * blockDim.x + threadIdx.x;
  if (idx >= K * Ncol) return;
  int k = idx / Ncol, n = idx % Ncol;
  Wt[(size_t)n * K + k] = f2bf(W[idx]);
}

// ---- LayerNorm over D=256, one row per 256-thread block -------------------
__global__ __launch_bounds__(256) void layernorm_k(
    const float* __restrict__ in, const float* __restrict__ g,
    const float* __restrict__ be, float* __restrict__ outF,
    unsigned short* __restrict__ outB) {
  __shared__ float red[256];
  int row = blockIdx.x, t = threadIdx.x;
  float v = in[(size_t)row * kD + t];
  red[t] = v;
  __syncthreads();
#pragma unroll
  for (int s = 128; s > 0; s >>= 1) {
    if (t < s) red[t] += red[t + s];
    __syncthreads();
  }
  float mean = red[0] * (1.0f / kD);
  __syncthreads();
  float dv = v - mean;
  red[t] = dv * dv;
  __syncthreads();
#pragma unroll
  for (int s = 128; s > 0; s >>= 1) {
    if (t < s) red[t] += red[t + s];
    __syncthreads();
  }
  float var = red[0] * (1.0f / kD);
  float o = dv * rsqrtf(var + kEPS) * g[t] + be[t];
  outF[(size_t)row * kD + t] = o;
  if (outB) outB[(size_t)row * kD + t] = f2bf(o);
}

// ---- bf16 GEMM, TDM-staged: C = A[M,K] x Bt[N,K]^T, fp32 accumulate -------
// 256 threads = 8 waves (4x2); block tile 128(M) x 64(N); each wave owns a
// 32x32 output tile -> 8 v_wmma per K-64 chunk. Double-buffered LDS tiles
// filled by wave 0 via tensor_load_to_lds; one s_barrier per chunk.
__global__ __launch_bounds__(256) void gemm_bf16_tdm(
    const unsigned short* __restrict__ A, int lda,
    const unsigned short* __restrict__ Bt, int ldb,
    const float* __restrict__ bias, const float* __restrict__ resid,
    float* __restrict__ outF, unsigned short* __restrict__ outB,
    int ldc, int Mtot, int Ntot, int K, int relu) {
  __shared__ unsigned short smem[2 * kBufElems];   // 54 KB
  const unsigned lds_base = (unsigned)(size_t)smem; // low 32b = LDS byte addr

  int w = threadIdx.x >> 5;
  int lane = threadIdx.x & 31;
  int mblk = blockIdx.y * 128;
  int nblk = blockIdx.x * 64;
  int m0 = (w >> 1) * 32;
  int n0 = (w & 1) * 32;
  const int steps = K >> 6;                        // K-chunks of 64

  if (w == 0) {
    tdm_load_2d(A + (size_t)mblk * lda, lds_base, 128, K, Mtot, lda);
    tdm_load_2d(Bt + (size_t)nblk * ldb, lds_base + 2 * kAtileElems,
                64, K, Ntot, ldb);
  }

  v8f c00 = {}, c01 = {}, c10 = {}, c11 = {};
  for (int s = 0; s < steps; ++s) {
    wait_tensorcnt0();
    __syncthreads();   // buffer s%2 ready; everyone done reading buffer (s+1)%2
    int cur = s & 1;
    if (w == 0 && s + 1 < steps) {
      int kc = (s + 1) << 6;
      unsigned lb = lds_base + (unsigned)((s + 1) & 1) * (2 * kBufElems);
      tdm_load_2d(A + (size_t)mblk * lda + kc, lb, 128, K, Mtot, lda);
      tdm_load_2d(Bt + (size_t)nblk * ldb + kc, lb + 2 * kAtileElems,
                  64, K, Ntot, ldb);
    }
    const unsigned short* As = smem + cur * kBufElems;
    const unsigned short* Bs = As + kAtileElems;
#pragma unroll
    for (int kk = 0; kk < 64; kk += 32) {
      v16bf a0 = lds_frag(As, m0, kk);
      v16bf a1 = lds_frag(As, m0 + 16, kk);
      v16bf b0 = lds_frag(Bs, n0, kk);
      v16bf b1 = lds_frag(Bs, n0 + 16, kk);
      c00 = __builtin_amdgcn_wmma_f32_16x16x32_bf16(false, a0, false, b0,
                                                    (short)0, c00, false, false);
      c01 = __builtin_amdgcn_wmma_f32_16x16x32_bf16(false, a0, false, b1,
                                                    (short)0, c01, false, false);
      c10 = __builtin_amdgcn_wmma_f32_16x16x32_bf16(false, a1, false, b0,
                                                    (short)0, c10, false, false);
      c11 = __builtin_amdgcn_wmma_f32_16x16x32_bf16(false, a1, false, b1,
                                                    (short)0, c11, false, false);
    }
  }

  int nr = lane & 15;
  int hsc = (lane & 16) ? 8 : 0;
  auto emit = [&](v8f acc, int mb, int nb) {
    AccF ac; ac.v = acc;
#pragma unroll
    for (int r = 0; r < 8; ++r) {
      int m = mb + r + hsc;
      int n = nb + nr;
      float v = ac.f[r];
      if (bias)  v += bias[n];
      if (relu)  v = fmaxf(v, 0.0f);
      size_t idx = (size_t)m * ldc + n;
      if (resid) v += resid[idx];
      if (outF)  outF[idx] = v;
      if (outB)  outB[idx] = f2bf(v);
    }
  };
  emit(c00, mblk + m0,      nblk + n0);
  emit(c01, mblk + m0,      nblk + n0 + 16);
  emit(c10, mblk + m0 + 16, nblk + n0);
  emit(c11, mblk + m0 + 16, nblk + n0 + 16);
}

// ---- triangle scores: per (b,h,a) a 64x32x64 GEMM, K=32 = one wmma --------
// scores[b,h,a,x,y] = (1/sqrt(DK)) * sum_d lk[b,x,a,h,d] * rk[b,a,y,h,d]
__global__ __launch_bounds__(128) void tri_scores(
    const unsigned short* __restrict__ lk, const unsigned short* __restrict__ rk,
    float* __restrict__ scores) {
  int w = threadIdx.x >> 5;
  int lane = threadIdx.x & 31;
  int a = blockIdx.x % kN;
  int h = (blockIdx.x / kN) % kH;
  int b = blockIdx.x / (kN * kH);
  const size_t bNNHD = (size_t)kN * kN * kH * kDK;
  const unsigned short* Abase = lk + (size_t)b * bNNHD + (size_t)a * kH * kDK + h * kDK;
  const unsigned short* Bbase = rk + (size_t)b * bNNHD + (size_t)a * kN * kH * kDK + h * kDK;
  float* sbase = scores + ((((size_t)b * kH + h) * kN + a) * kN) * kN;

  int m0 = w * 16;
  v16bf afr = load_frag(Abase, kN * kH * kDK, m0, 0);
  int nr = lane & 15;
  int hsc = (lane & 16) ? 8 : 0;
#pragma unroll
  for (int nt = 0; nt < 4; ++nt) {
    int n0 = nt * 16;
    v16bf bfr = load_frag(Bbase, kH * kDK, n0, 0);
    v8f c = {};
    c = __builtin_amdgcn_wmma_f32_16x16x32_bf16(false, afr, false, bfr, (short)0,
                                                c, false, false);
    AccF ac; ac.v = c;
#pragma unroll
    for (int r = 0; r < 8; ++r) {
      int x = m0 + r + hsc;
      int y = n0 + nr;
      sbase[(size_t)x * kN + y] = ac.f[r] * kSCALE;
    }
  }
}

// ---- masked softmax over intermediate node a (axis 2) ---------------------
__global__ __launch_bounds__(256) void softmax_mask(
    float* __restrict__ scores, const unsigned char* __restrict__ mask) {
  int idx = blockIdx.x * 256 + threadIdx.x;     // over (b,h,x,y)
  int y = idx % kN;
  int x = (idx / kN) % kN;
  int h = (idx / (kN * kN)) % kH;
  int b = idx / (kN * kN * kH);
  size_t base = (((size_t)b * kH + h) * kN) * (size_t)kN * kN + (size_t)x * kN + y;
  const size_t astr = (size_t)kN * kN;
  size_t mbase = ((size_t)b * kN + x) * (size_t)kN * kN + y;
  float s[kN];
  float mx = -3.0e38f;
#pragma unroll
  for (int a = 0; a < kN; ++a) {
    float v = scores[base + a * astr];
    if (mask[mbase + (size_t)a * kN]) v = kNEG;
    s[a] = v;
    mx = fmaxf(mx, v);
  }
  float sum = 0.0f;
#pragma unroll
  for (int a = 0; a < kN; ++a) {
    float e = __expf(s[a] - mx);
    s[a] = e;
    sum += e;
  }
  float inv = 1.0f / sum;
#pragma unroll
  for (int a = 0; a < kN; ++a) scores[base + a * astr] = s[a] * inv;
}

// ---- AV: x[b,x,y,h,d] = sum_a att[b,h,a,x,y]*lv[b,x,a,h,d]*rv[b,a,y,h,d] --
// One wave per (b,h,x,y); lane = d (DK==32==wave32).
__global__ __launch_bounds__(256) void av_kernel(
    const float* __restrict__ att, const unsigned short* __restrict__ lv,
    const unsigned short* __restrict__ rv, unsigned short* __restrict__ xbuf) {
  int gw = blockIdx.x * 8 + (threadIdx.x >> 5);
  int d = threadIdx.x & 31;
  int y = gw % kN;
  int x = (gw / kN) % kN;
  int h = (gw / (kN * kN)) % kH;
  int b = gw / (kN * kN * kH);
  const size_t bNNHD = (size_t)kN * kN * kH * kDK;
  const float* ab = att + (((size_t)b * kH + h) * kN) * (size_t)kN * kN +
                    (size_t)x * kN + y;                                  // + a*N*N
  const unsigned short* lvb = lv + (size_t)b * bNNHD +
                              (size_t)x * kN * kH * kDK + h * kDK + d;   // + a*H*DK
  const unsigned short* rvb = rv + (size_t)b * bNNHD +
                              (size_t)y * kH * kDK + h * kDK + d;        // + a*N*H*DK
  float acc = 0.0f;
#pragma unroll 4
  for (int a = 0; a < kN; ++a) {
    acc += ab[(size_t)a * kN * kN] * bf2f(lvb[(size_t)a * kH * kDK]) *
           bf2f(rvb[(size_t)a * kN * kH * kDK]);
  }
  xbuf[(((size_t)b * kN + x) * kN + y) * kD + h * kDK + d] = f2bf(acc);
}

// ---------------------------------------------------------------------------
extern "C" void kernel_launch(void* const* d_in, const int* in_sizes, int n_in,
                              void* d_out, int out_size, void* d_ws, size_t ws_size,
                              hipStream_t stream) {
  (void)in_sizes; (void)n_in; (void)out_size; (void)ws_size;
  const float* bg   = (const float*)d_in[0];
  const unsigned char* mask = (const unsigned char*)d_in[1];
  const float* Wlk  = (const float*)d_in[2];
  const float* Wrk  = (const float*)d_in[3];
  const float* Wlv  = (const float*)d_in[4];
  const float* Wrv  = (const float*)d_in[5];
  const float* Wo   = (const float*)d_in[6];
  const float* W1   = (const float*)d_in[7];
  const float* b1   = (const float*)d_in[8];
  const float* W2   = (const float*)d_in[9];
  const float* b2   = (const float*)d_in[10];
  const float* g1   = (const float*)d_in[11];
  const float* be1  = (const float*)d_in[12];
  const float* g2   = (const float*)d_in[13];
  const float* be2  = (const float*)d_in[14];
  float* out = (float*)d_out;

  // ---- workspace carve-up (~90 MB; everything L2-resident) ----
  char* w = (char*)d_ws;
  auto take = [&](size_t bytes) { char* p = w; w += (bytes + 255) & ~size_t(255); return p; };
  float*          hF   = (float*)take((size_t)kM * kD * 4);
  unsigned short* hB   = (unsigned short*)take((size_t)kM * kD * 2);
  unsigned short* lkB  = (unsigned short*)take((size_t)kM * kD * 2);
  unsigned short* rkB  = (unsigned short*)take((size_t)kM * kD * 2);
  unsigned short* lvB  = (unsigned short*)take((size_t)kM * kD * 2);
  unsigned short* rvB  = (unsigned short*)take((size_t)kM * kD * 2);
  float*          scr  = (float*)take((size_t)kB * kH * kN * kN * kN * 4);
  unsigned short* xbuf = (unsigned short*)take((size_t)kM * kD * 2);
  float*          hr   = (float*)take((size_t)kM * kD * 4);
  float*          h2F  = (float*)take((size_t)kM * kD * 4);
  unsigned short* h2B  = (unsigned short*)take((size_t)kM * kD * 2);
  unsigned short* f1   = (unsigned short*)take((size_t)kM * kDFF * 2);
  unsigned short* WlkT = (unsigned short*)take((size_t)kD * kD * 2);
  unsigned short* WrkT = (unsigned short*)take((size_t)kD * kD * 2);
  unsigned short* WlvT = (unsigned short*)take((size_t)kD * kD * 2);
  unsigned short* WrvT = (unsigned short*)take((size_t)kD * kD * 2);
  unsigned short* WoT  = (unsigned short*)take((size_t)kD * kD * 2);
  unsigned short* W1T  = (unsigned short*)take((size_t)kD * kDFF * 2);
  unsigned short* W2T  = (unsigned short*)take((size_t)kDFF * kD * 2);

  // ---- weights: fp32 [K,N] -> bf16 [N,K] ----
  int nDD = kD * kD;
  wconv_t<<<(nDD + 255) / 256, 256, 0, stream>>>(Wlk, WlkT, kD, kD);
  wconv_t<<<(nDD + 255) / 256, 256, 0, stream>>>(Wrk, WrkT, kD, kD);
  wconv_t<<<(nDD + 255) / 256, 256, 0, stream>>>(Wlv, WlvT, kD, kD);
  wconv_t<<<(nDD + 255) / 256, 256, 0, stream>>>(Wrv, WrvT, kD, kD);
  wconv_t<<<(nDD + 255) / 256, 256, 0, stream>>>(Wo,  WoT,  kD, kD);
  int nDF = kD * kDFF;
  wconv_t<<<(nDF + 255) / 256, 256, 0, stream>>>(W1, W1T, kD, kDFF);
  wconv_t<<<(nDF + 255) / 256, 256, 0, stream>>>(W2, W2T, kDFF, kD);

  // ---- LN1 ----
  layernorm_k<<<kM, 256, 0, stream>>>(bg, g1, be1, hF, hB);

  // ---- four projections: [8192,256] x [256,256] ----
  dim3 gP(kD / 64, kM / 128);
  gemm_bf16_tdm<<<gP, 256, 0, stream>>>(hB, kD, WlkT, kD, nullptr, nullptr,
                                        nullptr, lkB, kD, kM, kD, kD, 0);
  gemm_bf16_tdm<<<gP, 256, 0, stream>>>(hB, kD, WrkT, kD, nullptr, nullptr,
                                        nullptr, rkB, kD, kM, kD, kD, 0);
  gemm_bf16_tdm<<<gP, 256, 0, stream>>>(hB, kD, WlvT, kD, nullptr, nullptr,
                                        nullptr, lvB, kD, kM, kD, kD, 0);
  gemm_bf16_tdm<<<gP, 256, 0, stream>>>(hB, kD, WrvT, kD, nullptr, nullptr,
                                        nullptr, rvB, kD, kM, kD, kD, 0);

  // ---- triangle attention ----
  tri_scores<<<kB * kH * kN, 128, 0, stream>>>(lkB, rkB, scr);
  softmax_mask<<<(kB * kH * kN * kN) / 256, 256, 0, stream>>>(scr, mask);
  av_kernel<<<(kB * kH * kN * kN) / 8, 256, 0, stream>>>(scr, lvB, rvB, xbuf);

  // ---- output projection + residual (residual uses normed tensor hF) ----
  gemm_bf16_tdm<<<gP, 256, 0, stream>>>(xbuf, kD, WoT, kD, nullptr, hF,
                                        hr, nullptr, kD, kM, kD, kD, 0);

  // ---- LN2 ----
  layernorm_k<<<kM, 256, 0, stream>>>(hr, g2, be2, h2F, h2B);

  // ---- FFN ----
  dim3 gF1(kDFF / 64, kM / 128);
  gemm_bf16_tdm<<<gF1, 256, 0, stream>>>(h2B, kD, W1T, kD, b1, nullptr,
                                         nullptr, f1, kDFF, kM, kDFF, kD, 1);
  gemm_bf16_tdm<<<gP, 256, 0, stream>>>(f1, kDFF, W2T, kDFF, b2, h2F,
                                        out, nullptr, kD, kM, kD, kDFF, 0);
}
